// MinimalSSM_12438225289387
// MI455X (gfx1250) — matrix operand order repaired and checked by reading, not compiled, not run on hardware
//
#include <hip/hip_runtime.h>
#include <hip/hip_bf16.h>

typedef __attribute__((ext_vector_type(16))) __bf16 v16bf;
typedef __attribute__((ext_vector_type(8)))  __bf16 v8bf;
typedef __attribute__((ext_vector_type(8)))  float  v8f;
typedef __attribute__((ext_vector_type(4)))  int    v4i_;

#define D_MODEL 1024
#define D_STATE 16
#define NPROJ   (D_MODEL + 2 * D_STATE)   // 1056
#define BATCH   4
#define SEQ     2048
#define MTOT    (BATCH * SEQ)             // 8192

// --------------------------------------------------------------------------
// CDNA5 async global->LDS copy (ASYNCcnt path, CDNA5 ISA)
// Signature (from compiler diagnostic): (v4i AS1*, v4i AS3*, Ii off, Ii cpol)
// --------------------------------------------------------------------------
#if __has_builtin(__builtin_amdgcn_global_load_async_to_lds_b128) && \
    __has_builtin(__builtin_amdgcn_s_wait_asynccnt)
#define USE_ASYNC_LDS 1
#else
#define USE_ASYNC_LDS 0
#endif

#if USE_ASYNC_LDS
__device__ __forceinline__ void async_cp16(const __bf16* g, __bf16* l) {
  // AS1 pointer = same 64-bit VA; AS3 pointer = low 32 bits of generic VA
  // (flat aperture).  Integer-mediated casts keep
  // this legal C++ regardless of clang's addrspacecast rules.
  __builtin_amdgcn_global_load_async_to_lds_b128(
      (__attribute__((address_space(1))) v4i_*)(unsigned long long)(size_t)g,
      (__attribute__((address_space(3))) v4i_*)(unsigned)(size_t)l,
      0, 0);
}
__device__ __forceinline__ void wait_async0() {
  __builtin_amdgcn_s_wait_asynccnt(0);
}
#endif

// ---------------------------------------------------------------------------
// fp32 -> bf16 elementwise
// ---------------------------------------------------------------------------
__global__ __launch_bounds__(256) void cvt_f32_bf16(const float* __restrict__ in,
                                                    __bf16* __restrict__ out,
                                                    int count) {
  int i = blockIdx.x * blockDim.x + threadIdx.x;
  int stride = gridDim.x * blockDim.x;
  for (; i < count; i += stride) out[i] = (__bf16)in[i];
}

// ---------------------------------------------------------------------------
// W [K][N] fp32 row-major  ->  Wt [N][K] bf16 (transposed)
// ---------------------------------------------------------------------------
__global__ __launch_bounds__(256) void transpose_bf16(const float* __restrict__ W,
                                                      __bf16* __restrict__ Wt,
                                                      int K, int N) {
  int i = blockIdx.x * blockDim.x + threadIdx.x;
  int total = K * N;
  int stride = gridDim.x * blockDim.x;
  for (; i < total; i += stride) {
    int k = i / N;
    int n = i - k * N;
    Wt[(size_t)n * K + k] = (__bf16)W[i];
  }
}

// ---------------------------------------------------------------------------
// bf16 WMMA GEMM:  C[M][N] = A[M][K] * Bt[N][K]^T + bias[N]   (fp32 out)
// block tile 128x128, BK=64, double-buffered LDS, async global->LDS DMA.
// 8 waves (2m x 4n); each wave 64x32 = 4x2 WMMA tiles; 16 WMMA / k-iter.
// ---------------------------------------------------------------------------
#define BM 128
#define BN 128
#define BKK 64
#define LDT 72   // padded LDS row stride (64 data + 8 pad bf16) = 144B = 36 dw

__global__ __launch_bounds__(256) void gemm_bf16_wmma(
    const __bf16* __restrict__ A,    // [M][K]
    const __bf16* __restrict__ Bt,   // [N][K]
    const float*  __restrict__ bias, // [N]
    float* __restrict__ C,           // [M][N]
    int M, int N, int K) {
  __shared__ __bf16 sA[2][BM * LDT];   // 2 x 18 KB
  __shared__ __bf16 sB[2][BN * LDT];   // 2 x 18 KB

  const int tid  = threadIdx.x;
  const int lane = tid & 31;
  const int wid  = tid >> 5;
  const int half = lane >> 4;   // 0: lanes 0-15, 1: lanes 16-31
  const int lrow = lane & 15;
  const int wm   = wid >> 2;    // 0..1  -> 64 rows
  const int wn   = wid & 3;     // 0..3  -> 32 cols

  const int bm = blockIdx.y * BM;
  const int bn = blockIdx.x * BN;

  // cooperative loader: thread t covers row lr, 32 bf16 (64B) per tile
  const int lr = tid >> 1;          // 0..127
  const int lc = (tid & 1) << 5;    // 0 or 32
  const int  gbn = bn + lr;
  const bool bok = (gbn < N);

  v8f acc[4][2] = {};

  auto load_tiles = [&](int k0, int buf) {
    const __bf16* ga = A + (size_t)(bm + lr) * K + (k0 + lc);
    __bf16* la = &sA[buf][lr * LDT + lc];
#if USE_ASYNC_LDS
    async_cp16(ga, la);
    async_cp16(ga + 8, la + 8);
    async_cp16(ga + 16, la + 16);
    async_cp16(ga + 24, la + 24);
#else
    *(v8bf*)la        = *(const v8bf*)ga;
    *(v8bf*)(la + 8)  = *(const v8bf*)(ga + 8);
    *(v8bf*)(la + 16) = *(const v8bf*)(ga + 16);
    *(v8bf*)(la + 24) = *(const v8bf*)(ga + 24);
#endif
    __bf16* lb = &sB[buf][lr * LDT + lc];
    if (bok) {
      const __bf16* gb = Bt + (size_t)gbn * K + (k0 + lc);
#if USE_ASYNC_LDS
      async_cp16(gb, lb);
      async_cp16(gb + 8, lb + 8);
      async_cp16(gb + 16, lb + 16);
      async_cp16(gb + 24, lb + 24);
#else
      *(v8bf*)lb        = *(const v8bf*)gb;
      *(v8bf*)(lb + 8)  = *(const v8bf*)(gb + 8);
      *(v8bf*)(lb + 16) = *(const v8bf*)(gb + 16);
      *(v8bf*)(lb + 24) = *(const v8bf*)(gb + 24);
#endif
    } else {
      v8bf z = {};
      *(v8bf*)lb        = z;
      *(v8bf*)(lb + 8)  = z;
      *(v8bf*)(lb + 16) = z;
      *(v8bf*)(lb + 24) = z;
    }
  };

  const int nk = K / BKK;   // 16 for K=1024
  load_tiles(0, 0);

  for (int i = 0; i < nk; ++i) {
#if USE_ASYNC_LDS
    wait_async0();            // this wave's DMA into buf (i&1) complete
#endif
    __syncthreads();          // block-wide: current buffer ready
    if (i + 1 < nk) load_tiles((i + 1) * BKK, (i + 1) & 1);  // overlap w/ WMMA

    const __bf16* cA = &sA[i & 1][0];
    const __bf16* cB = &sB[i & 1][0];
#pragma unroll
    for (int ks = 0; ks < 2; ++ks) {
      // B frags: lane<16 K=0..15 of col N=lrow; lane>=16 K=16..31
      v16bf bfrag[2];
#pragma unroll
      for (int ni = 0; ni < 2; ++ni) {
        const int nn = wn * 32 + ni * 16 + lrow;
        v8bf blo = *(const v8bf*)&cB[nn * LDT + ks * 32 + half * 16];
        v8bf bhi = *(const v8bf*)&cB[nn * LDT + ks * 32 + half * 16 + 8];
        bfrag[ni] = __builtin_shufflevector(blo, bhi,
            0, 1, 2, 3, 4, 5, 6, 7, 8, 9, 10, 11, 12, 13, 14, 15);
      }
#pragma unroll
      for (int mi = 0; mi < 4; ++mi) {
        // A frag: lane<16 row M=lrow holds K=0..15; lane>=16 K=16..31
        const int mm = wm * 64 + mi * 16 + lrow;
        v8bf alo = *(const v8bf*)&cA[mm * LDT + ks * 32 + half * 16];
        v8bf ahi = *(const v8bf*)&cA[mm * LDT + ks * 32 + half * 16 + 8];
        v16bf afrag = __builtin_shufflevector(alo, ahi,
            0, 1, 2, 3, 4, 5, 6, 7, 8, 9, 10, 11, 12, 13, 14, 15);
#pragma unroll
        for (int ni = 0; ni < 2; ++ni) {
          acc[mi][ni] = __builtin_amdgcn_wmma_f32_16x16x32_bf16(
              false, afrag, false, bfrag[ni], (short)0, acc[mi][ni], false, false);
        }
      }
    }
    __syncthreads();          // all waves done reading buf (i&1) before reuse
  }

  // epilogue: C/D layout -> VGPR v: lanes 0-15 M=v, lanes 16-31 M=v+8; N=lrow
#pragma unroll
  for (int ni = 0; ni < 2; ++ni) {
    const int gn = bn + wn * 32 + ni * 16 + lrow;
    if (gn < N) {
      const float bv = bias[gn];
#pragma unroll
      for (int mi = 0; mi < 4; ++mi) {
#pragma unroll
        for (int v = 0; v < 8; ++v) {
          const int gm = bm + wm * 64 + mi * 16 + half * 8 + v;
          C[(size_t)gm * N + gn] = acc[mi][ni][v] + bv;
        }
      }
    }
  }
}

// ---------------------------------------------------------------------------
// Fused selective-scan: softplus(delta), h = exp(delta*A)*h + delta*B*x,
// y = <h,C> + D*x.  Lanes carry the n=16 state dim (2 channels/wave32),
// 4-step __shfl_xor butterfly for the n-reduction.  y staged in LDS as
// 16x16 tiles so global bf16 stores are coalesced (feeds GEMM2).
// ---------------------------------------------------------------------------
__global__ __launch_bounds__(256) void ssm_scan(
    const float* __restrict__ proj,   // [MTOT][NPROJ]
    const float* __restrict__ x,      // [MTOT][D_MODEL]
    const float* __restrict__ A_log,  // [D_MODEL][D_STATE]
    const float* __restrict__ Dp,     // [D_MODEL]
    __bf16* __restrict__ yb) {        // [MTOT][D_MODEL] bf16
  const int tid  = threadIdx.x;
  const int lane = tid & 31;
  const int wid  = tid >> 5;
  const int sub  = lane >> 4;     // which of the wave's 2 channels
  const int n    = lane & 15;     // state index
  const int c    = wid * 2 + sub; // channel within block (0..15)
  const int b    = blockIdx.y;
  const int d0   = blockIdx.x * 16;
  const int d    = d0 + c;

  __shared__ __bf16 sy[16][16];

  const float An = -__expf(A_log[d * D_STATE + n]);
  const float Dv = Dp[d];
  float h = 0.f;

  const size_t base = (size_t)b * SEQ;

  for (int l = 0; l < SEQ; ++l) {
    const size_t row = base + l;
    const float* pr = proj + row * NPROJ;
    const float draw = pr[d];
    // numerically-stable softplus
    const float delta = (draw > 0.f) ? (draw + log1pf(__expf(-draw)))
                                     : log1pf(__expf(draw));
    const float xv = x[row * D_MODEL + d];
    const float Bn = pr[D_MODEL + n];
    const float Cn = pr[D_MODEL + D_STATE + n];

    h = __expf(delta * An) * h + (delta * Bn) * xv;

    float p = h * Cn;
    p += __shfl_xor(p, 1, 32);
    p += __shfl_xor(p, 2, 32);
    p += __shfl_xor(p, 4, 32);
    p += __shfl_xor(p, 8, 32);
    const float yv = p + Dv * xv;

    const int phase = l & 15;
    if (n == phase) sy[phase][c] = (__bf16)yv;

    if (phase == 15) {
      __syncthreads();
      const int s  = tid >> 4;  // step within 16-step window
      const int ch = tid & 15;  // channel
      yb[(base + (size_t)(l - 15 + s)) * D_MODEL + d0 + ch] = sy[s][ch];
      __syncthreads();
    }
  }
}

// ---------------------------------------------------------------------------
extern "C" void kernel_launch(void* const* d_in, const int* in_sizes, int n_in,
                              void* d_out, int out_size, void* d_ws, size_t ws_size,
                              hipStream_t stream) {
  (void)in_sizes; (void)n_in; (void)out_size; (void)ws_size;
  const float* x       = (const float*)d_in[0];
  const float* A_log   = (const float*)d_in[1];
  const float* Dp      = (const float*)d_in[2];
  const float* W_xproj = (const float*)d_in[3];
  const float* b_xproj = (const float*)d_in[4];
  const float* W_out   = (const float*)d_in[5];
  const float* b_out   = (const float*)d_in[6];
  float* out = (float*)d_out;

  char* ws = (char*)d_ws;
  size_t off = 0;
  auto wsalloc = [&](size_t bytes) {
    void* p = ws + off;
    off = (off + bytes + 255) & ~(size_t)255;
    return p;
  };
  __bf16* xb   = (__bf16*)wsalloc((size_t)MTOT * D_MODEL * 2);    // 16 MB
  __bf16* Wxt  = (__bf16*)wsalloc((size_t)NPROJ * D_MODEL * 2);   // 2.1 MB
  __bf16* Wot  = (__bf16*)wsalloc((size_t)D_MODEL * D_MODEL * 2); // 2 MB
  float*  proj = (float*) wsalloc((size_t)MTOT * NPROJ * 4);      // 34.6 MB
  __bf16* yb   = (__bf16*)wsalloc((size_t)MTOT * D_MODEL * 2);    // 16 MB

  // 1) x -> bf16
  {
    int count = MTOT * D_MODEL;
    cvt_f32_bf16<<<(count + 255) / 256, 256, 0, stream>>>(x, xb, count);
  }
  // 2) weights -> transposed bf16
  transpose_bf16<<<(D_MODEL * NPROJ + 255) / 256, 256, 0, stream>>>(
      W_xproj, Wxt, D_MODEL, NPROJ);
  transpose_bf16<<<(D_MODEL * D_MODEL + 255) / 256, 256, 0, stream>>>(
      W_out, Wot, D_MODEL, D_MODEL);
  // 3) proj = x @ W_xproj + b_xproj
  gemm_bf16_wmma<<<dim3((NPROJ + BN - 1) / BN, MTOT / BM), 256, 0, stream>>>(
      xb, Wxt, b_xproj, proj, MTOT, NPROJ, D_MODEL);
  // 4) fused scan -> y (bf16)
  ssm_scan<<<dim3(D_MODEL / 16, BATCH), 256, 0, stream>>>(proj, x, A_log, Dp, yb);
  // 5) out = y @ W_out + b_out
  gemm_bf16_wmma<<<dim3(D_MODEL / BN, MTOT / BM), 256, 0, stream>>>(
      yb, Wot, b_out, out, MTOT, D_MODEL, D_MODEL);
}
